// DcMoeBlock_8400956031337
// MI455X (gfx1250) — matrix operand order repaired and checked by reading, not compiled, not run on hardware
//
#include <hip/hip_runtime.h>
#include <math.h>

typedef __bf16 bf16;
typedef __attribute__((ext_vector_type(16))) __bf16 v16bf;
typedef __attribute__((ext_vector_type(8)))  __bf16 v8bf;
typedef __attribute__((ext_vector_type(8)))  float  v8f;
typedef __attribute__((ext_vector_type(4)))  float  v4f;
typedef unsigned int u32;
typedef unsigned long long u64;
typedef __attribute__((ext_vector_type(4))) unsigned int u32x4;
typedef __attribute__((ext_vector_type(8))) int i32x8;
typedef __attribute__((ext_vector_type(4))) int i32x4;

#define GN   4       // groups
#define TPG  2048    // tokens per group
#define EMBD 1024
#define MLPD 4096
#define NE   8       // experts
#define TOPK 2
#define CAP  640     // ceil(TPG * 1.25 * K / E)

// ---------------------------------------------------------------------------
// TDM: build a 2-D tensor-DMA descriptor (D#) and issue TENSOR_LOAD_TO_LDS.
// data_size = 2 bytes (bf16). Fields per CDNA5 ISA §8 (CDNA5 ISA).
// ---------------------------------------------------------------------------
__device__ __forceinline__ void tdm_load_2d_bf16(
    u32 lds_off, u64 gaddr, u32 tensor_d0, u32 tensor_d1, u64 stride0,
    u32 tile_d0, u32 tile_d1) {
    // group0: [1:0]=count=1 | [63:32]=lds_addr | [120:64]=global_addr | [127:126]=type=2
    u32x4 g0 = {
        1u,
        lds_off,
        (u32)(gaddr & 0xffffffffull),
        (u32)((gaddr >> 32) & 0x01ffffffull) | (2u << 30)
    };
    // group1: data_size=1 (2B) @ [17:16]; tensor_dim0 @ [79:48]; tensor_dim1 @ [111:80];
    // tile_dim0 @ [127:112]; tile_dim1 @ [143:128]; tensor_dim0_stride @ [207:160]
    i32x8 g1 = {
        (int)(1u << 16),
        (int)((tensor_d0 & 0xffffu) << 16),
        (int)((tensor_d0 >> 16) | ((tensor_d1 & 0xffffu) << 16)),
        (int)((tensor_d1 >> 16) | ((tile_d0 & 0xffffu) << 16)),
        (int)(tile_d1 & 0xffffu),
        (int)(u32)(stride0 & 0xffffffffull),
        (int)(u32)((stride0 >> 32) & 0xffffull),
        0
    };
    i32x4 z4 = {};
#if defined(__clang_major__) && (__clang_major__ >= 23)
    i32x8 z8 = {};
    __builtin_amdgcn_tensor_load_to_lds(g0, g1, z4, z4, z8, 0);
#else
    __builtin_amdgcn_tensor_load_to_lds(g0, g1, z4, z4, 0);
#endif
}

// ---------------------------------------------------------------------------
// init: slot_token = -1, stats = 0
// ---------------------------------------------------------------------------
__global__ void moe_init(int* __restrict__ slot_token, float* __restrict__ stats) {
    int i = blockIdx.x * 256 + threadIdx.x;
    if (i < GN * NE * CAP) slot_token[i] = -1;
    if (i < 2 * GN * NE + 1) stats[i] = 0.0f;
}

// ---------------------------------------------------------------------------
// router: logits, softmax, top-2, stats. one wave32 per token, 8 waves/block.
// ---------------------------------------------------------------------------
__global__ __launch_bounds__(256) void moe_router(
    const float* __restrict__ x, const float* __restrict__ pad,
    const float* __restrict__ rk,
    int* __restrict__ tok_e, float* __restrict__ tok_gate,
    float* __restrict__ stats /* [G*NE probsum][G*NE toksum][1 zacc] */) {
    int g    = blockIdx.y;
    int wave = threadIdx.x >> 5;
    int lane = threadIdx.x & 31;
    int t    = blockIdx.x * 8 + wave;

    __shared__ float sprob[NE];
    __shared__ float stok[NE];
    __shared__ float sz[1];
    if (threadIdx.x < NE) { sprob[threadIdx.x] = 0.f; stok[threadIdx.x] = 0.f; }
    if (threadIdx.x == 0) sz[0] = 0.f;
    __syncthreads();

    float acc[NE];
#pragma unroll
    for (int e = 0; e < NE; e++) acc[e] = 0.f;

    const float* xp = x + ((size_t)g * TPG + t) * EMBD;
    for (int d = lane; d < EMBD; d += 32) {
        float xv = xp[d];
        const float* r = rk + (size_t)d * NE;
#pragma unroll
        for (int e = 0; e < NE; e++) acc[e] = fmaf(xv, r[e], acc[e]);
    }
#pragma unroll
    for (int e = 0; e < NE; e++) {
        for (int off = 16; off >= 1; off >>= 1) acc[e] += __shfl_xor(acc[e], off);
    }

    if (lane == 0) {
        float mx = acc[0];
        for (int e = 1; e < NE; e++) mx = fmaxf(mx, acc[e]);
        float p[NE];
        float se = 0.f;
        for (int e = 0; e < NE; e++) { p[e] = __expf(acc[e] - mx); se += p[e]; }
        float inv = 1.f / se;
        for (int e = 0; e < NE; e++) p[e] *= inv;
        float lse = mx + __logf(se);

        int i0 = 0;
        for (int e = 1; e < NE; e++) if (p[e] > p[i0]) i0 = e;
        int i1 = -1;
        for (int e = 0; e < NE; e++) if (e != i0 && (i1 < 0 || p[e] > p[i1])) i1 = e;

        float np = 1.f - pad[(size_t)g * TPG + t];
        size_t base = ((size_t)g * TPG + t) * TOPK;
        tok_e[base + 0]    = (np > 0.5f) ? i0 : -1;
        tok_e[base + 1]    = (np > 0.5f) ? i1 : -1;
        tok_gate[base + 0] = p[i0] * np;
        tok_gate[base + 1] = p[i1] * np;

        for (int e = 0; e < NE; e++) atomicAdd(&sprob[e], p[e]);
        atomicAdd(&stok[i0], 1.f);
        atomicAdd(&stok[i1], 1.f);
        atomicAdd(&sz[0], lse * lse);
    }
    __syncthreads();
    if (threadIdx.x < NE) {
        atomicAdd(&stats[g * NE + threadIdx.x], sprob[threadIdx.x]);
        atomicAdd(&stats[GN * NE + g * NE + threadIdx.x], stok[threadIdx.x]);
    }
    if (threadIdx.x == 0) atomicAdd(&stats[2 * GN * NE], sz[0]);
}

// ---------------------------------------------------------------------------
// dispatch: exact cumsum-order capacity assignment, serial per group.
// ---------------------------------------------------------------------------
__global__ void moe_dispatch(const int* __restrict__ tok_e,
                             int* __restrict__ tok_slot,
                             int* __restrict__ slot_token) {
    if (threadIdx.x != 0) return;
    int g = blockIdx.x;
    int cnt[NE];
#pragma unroll
    for (int e = 0; e < NE; e++) cnt[e] = 0;
    for (int t = 0; t < TPG; t++) {
        size_t base = ((size_t)g * TPG + t) * TOPK;
        for (int k = 0; k < TOPK; k++) {
            int e = tok_e[base + k];
            int c = -1;
            if (e >= 0) {
                int pos = cnt[e]++;
                if (pos < CAP) {
                    c = pos;
                    slot_token[((size_t)g * NE + e) * CAP + c] = t;
                }
            }
            tok_slot[base + k] = c;
        }
    }
}

// ---------------------------------------------------------------------------
// weight conversion fp32 -> bf16, transposed to K-contiguous layouts:
//   w0t/w1t: [e][h][m]  (GEMM1: K=m contiguous per output column h)
//   wot    : [e][m][h]  (GEMM2: K=h contiguous per output column m)
// ---------------------------------------------------------------------------
__global__ void moe_convert(const float* __restrict__ wi0, const float* __restrict__ wi1,
                            const float* __restrict__ wo,
                            bf16* __restrict__ w0t, bf16* __restrict__ w1t,
                            bf16* __restrict__ wot) {
    size_t N = (size_t)NE * MLPD * EMBD;
    for (size_t i = blockIdx.x * (size_t)blockDim.x + threadIdx.x; i < N;
         i += (size_t)gridDim.x * blockDim.x) {
        size_t m = i % EMBD;
        size_t h = (i / EMBD) % MLPD;
        size_t e = i / ((size_t)EMBD * MLPD);
        w0t[i] = (bf16)wi0[(e * EMBD + m) * MLPD + h];
        w1t[i] = (bf16)wi1[(e * EMBD + m) * MLPD + h];
        size_t h2 = i % MLPD;
        size_t m2 = (i / MLPD) % EMBD;
        wot[i] = (bf16)wo[(e * MLPD + h2) * EMBD + m2];
    }
}

// fragment union: one WMMA operand = 2x 16B LDS loads
union frag_t { v16bf v; v8bf h[2]; };

// ---------------------------------------------------------------------------
// GEMM1 fused gated FFN: per (g,e):  H = silu(X@wi1) * (X@wi0)
// M=CAP=640, N=MLP=4096, K=EMB=1024.  Block tile 64x128, 8 waves,
// each wave: 16x64 = 4 bf16 WMMA tiles x 2 weight matrices.
// A gathered from scattered tokens (fp32->bf16 ds_store path, scatter).
// B tiles DMA'd by the Tensor Data Mover, double-buffered in LDS.
// ---------------------------------------------------------------------------
__global__ __launch_bounds__(256) void moe_ffn1(
    const float* __restrict__ x, const int* __restrict__ slot_token,
    const bf16* __restrict__ w0t, const bf16* __restrict__ w1t,
    bf16* __restrict__ H) {
    __shared__ alignas(16) bf16 As [2][64 * 32];
    __shared__ alignas(16) bf16 Bs0[2][128 * 32];
    __shared__ alignas(16) bf16 Bs1[2][128 * 32];

    int ge = blockIdx.z;
    int g = ge / NE, e = ge % NE;
    int cBase = blockIdx.y * 64;
    int hBase = blockIdx.x * 128;
    int tid = threadIdx.x;
    int wave = tid >> 5, lane = tid & 31;
    int wm = wave & 3, wn = wave >> 2;

    // A-fill: thread -> (row, 8-col group)
    int ar = tid >> 2;
    int aq = (tid & 3) * 8;
    int tok = slot_token[((size_t)g * NE + e) * CAP + cBase + ar];
    const float* xrow = (tok >= 0) ? (x + ((size_t)g * TPG + tok) * EMBD) : nullptr;

    const u64 b0base = (u64)(size_t)(w0t + ((size_t)e * MLPD + hBase) * EMBD);
    const u64 b1base = (u64)(size_t)(w1t + ((size_t)e * MLPD + hBase) * EMBD);

    v8f acc0[4] = {};
    v8f acc1[4] = {};

    auto fillA = [&](int buf, int k0) {
        v8bf av = {};
        if (xrow) {
            const v4f* p = (const v4f*)(xrow + k0 + aq);
            v4f f0 = p[0], f1 = p[1];
#pragma unroll
            for (int j = 0; j < 4; j++) { av[j] = (bf16)f0[j]; av[4 + j] = (bf16)f1[j]; }
        }
        *(v8bf*)&As[buf][ar * 32 + aq] = av;
    };
    auto issueB = [&](int buf, int k0) {
        tdm_load_2d_bf16((u32)(size_t)&Bs0[buf][0], b0base + (u64)k0 * 2,
                         EMBD, MLPD, EMBD, 32, 128);
        tdm_load_2d_bf16((u32)(size_t)&Bs1[buf][0], b1base + (u64)k0 * 2,
                         EMBD, MLPD, EMBD, 32, 128);
    };

    const int NSTEP = EMBD / 32;
    fillA(0, 0);
    if (wave == 0) issueB(0, 0);

    for (int i = 0; i < NSTEP; i++) {
        int b = i & 1;
        __syncthreads();                       // prior compute on buf b^1 finished
        if (i + 1 < NSTEP) {
            fillA(b ^ 1, (i + 1) * 32);        // prefetch A (gather, ds_store)
            if (wave == 0) issueB(b ^ 1, (i + 1) * 32);  // prefetch B via TDM
        }
        if (wave == 0) {
            if (i + 1 < NSTEP) __builtin_amdgcn_s_wait_tensorcnt(2);  // buf b pair done
            else               __builtin_amdgcn_s_wait_tensorcnt(0);
        }
        __syncthreads();                       // buf b fills visible to all waves

        // A fragment (contiguous K layout: lane<16 -> K 0-15; lane>=16 -> K 16-31)
        int arow = wm * 16 + (lane & 15);
        int kba = (lane >> 4) * 16;
        frag_t fa;
        fa.h[0] = *(const v8bf*)&As[b][arow * 32 + kba];
        fa.h[1] = *(const v8bf*)&As[b][arow * 32 + kba + 8];

        // batch all B-fragment loads, then issue WMMAs back-to-back
        int kbb = (lane >> 4) * 16;
        frag_t fb0[4], fb1[4];
#pragma unroll
        for (int nt = 0; nt < 4; nt++) {
            int n = wn * 64 + nt * 16 + (lane & 15);
            fb0[nt].h[0] = *(const v8bf*)&Bs0[b][n * 32 + kbb];
            fb0[nt].h[1] = *(const v8bf*)&Bs0[b][n * 32 + kbb + 8];
            fb1[nt].h[0] = *(const v8bf*)&Bs1[b][n * 32 + kbb];
            fb1[nt].h[1] = *(const v8bf*)&Bs1[b][n * 32 + kbb + 8];
        }
#pragma unroll
        for (int nt = 0; nt < 4; nt++) {
            acc0[nt] = __builtin_amdgcn_wmma_f32_16x16x32_bf16(
                false, fa.v, false, fb0[nt].v, (short)0, acc0[nt], false, false);
            acc1[nt] = __builtin_amdgcn_wmma_f32_16x16x32_bf16(
                false, fa.v, false, fb1[nt].v, (short)0, acc1[nt], false, false);
        }
    }

    // epilogue: silu(h1) * h0 -> bf16 H
    int rrow  = cBase + wm * 16 + (lane >> 4) * 8;
    int ccol0 = hBase + wn * 64 + (lane & 15);
#pragma unroll
    for (int nt = 0; nt < 4; nt++) {
        int hcol = ccol0 + nt * 16;
#pragma unroll
        for (int i = 0; i < 8; i++) {
            float h0v = acc0[nt][i];
            float h1v = acc1[nt][i];
            float s = h1v / (1.f + __expf(-h1v));
            H[(((size_t)ge) * CAP + rrow + i) * MLPD + hcol] = (bf16)(s * h0v);
        }
    }
}

// ---------------------------------------------------------------------------
// GEMM2: exp_out = H @ wo.  M=640, N=EMB=1024, K=MLP=4096.
// Both A and B tiles are regular 2-D tiles -> both DMA'd by the TDM,
// double-buffered; waves only compute.
// ---------------------------------------------------------------------------
__global__ __launch_bounds__(256) void moe_ffn2(
    const bf16* __restrict__ H, const bf16* __restrict__ wot,
    float* __restrict__ eo) {
    __shared__ alignas(16) bf16 As[2][64 * 32];
    __shared__ alignas(16) bf16 Bs[2][128 * 32];

    int ge = blockIdx.z;
    int e = ge % NE;
    int cBase = blockIdx.y * 64;
    int mBase = blockIdx.x * 128;
    int tid = threadIdx.x;
    int wave = tid >> 5, lane = tid & 31;
    int wm = wave & 3, wn = wave >> 2;

    const u64 abase = (u64)(size_t)(H + ((size_t)ge * CAP + cBase) * MLPD);
    const u64 bbase = (u64)(size_t)(wot + ((size_t)e * EMBD + mBase) * MLPD);

    v8f acc[4] = {};

    auto issueAB = [&](int buf, int k0) {
        tdm_load_2d_bf16((u32)(size_t)&As[buf][0], abase + (u64)k0 * 2,
                         MLPD, CAP, MLPD, 32, 64);
        tdm_load_2d_bf16((u32)(size_t)&Bs[buf][0], bbase + (u64)k0 * 2,
                         MLPD, EMBD, MLPD, 32, 128);
    };

    const int NSTEP = MLPD / 32;
    if (wave == 0) issueAB(0, 0);

    for (int i = 0; i < NSTEP; i++) {
        int b = i & 1;
        __syncthreads();
        if (i + 1 < NSTEP && wave == 0) issueAB(b ^ 1, (i + 1) * 32);
        if (wave == 0) {
            if (i + 1 < NSTEP) __builtin_amdgcn_s_wait_tensorcnt(2);
            else               __builtin_amdgcn_s_wait_tensorcnt(0);
        }
        __syncthreads();

        int arow = wm * 16 + (lane & 15);
        int kba = (lane >> 4) * 16;
        frag_t fa;
        fa.h[0] = *(const v8bf*)&As[b][arow * 32 + kba];
        fa.h[1] = *(const v8bf*)&As[b][arow * 32 + kba + 8];

        int kbb = (lane >> 4) * 16;
        frag_t fb[4];
#pragma unroll
        for (int nt = 0; nt < 4; nt++) {
            int n = wn * 64 + nt * 16 + (lane & 15);
            fb[nt].h[0] = *(const v8bf*)&Bs[b][n * 32 + kbb];
            fb[nt].h[1] = *(const v8bf*)&Bs[b][n * 32 + kbb + 8];
        }
#pragma unroll
        for (int nt = 0; nt < 4; nt++) {
            acc[nt] = __builtin_amdgcn_wmma_f32_16x16x32_bf16(
                false, fa.v, false, fb[nt].v, (short)0, acc[nt], false, false);
        }
    }

    int rrow  = cBase + wm * 16 + (lane >> 4) * 8;
    int mcol0 = mBase + wn * 64 + (lane & 15);
#pragma unroll
    for (int nt = 0; nt < 4; nt++) {
        int mcol = mcol0 + nt * 16;
#pragma unroll
        for (int i = 0; i < 8; i++) {
            eo[(((size_t)ge) * CAP + rrow + i) * EMBD + mcol] = acc[nt][i];
        }
    }
}

// ---------------------------------------------------------------------------
// combine: out[g,t,:] = sum_k gate_k * exp_out[g, e_k, c_k, :]
// ---------------------------------------------------------------------------
__global__ __launch_bounds__(256) void moe_combine(
    const int* __restrict__ tok_e, const float* __restrict__ tok_gate,
    const int* __restrict__ tok_slot, const float* __restrict__ eo,
    float* __restrict__ out) {
    int gt = blockIdx.x;
    int g = gt / TPG;
    size_t base = (size_t)gt * TOPK;
    int e0 = tok_e[base], e1 = tok_e[base + 1];
    int c0 = tok_slot[base], c1 = tok_slot[base + 1];
    float g0 = tok_gate[base], g1 = tok_gate[base + 1];
    const float* p0 = (c0 >= 0) ? eo + (((size_t)g * NE + e0) * CAP + c0) * EMBD : nullptr;
    const float* p1 = (c1 >= 0) ? eo + (((size_t)g * NE + e1) * CAP + c1) * EMBD : nullptr;
    for (int m = threadIdx.x; m < EMBD; m += blockDim.x) {
        float v = 0.f;
        if (p0) v += g0 * p0[m];
        if (p1) v += g1 * p1[m];
        out[(size_t)gt * EMBD + m] = v;
    }
}

// ---------------------------------------------------------------------------
// finalize aux loss
// ---------------------------------------------------------------------------
__global__ void moe_finalize(const float* __restrict__ stats, float* __restrict__ aux) {
    if (threadIdx.x != 0 || blockIdx.x != 0) return;
    float lb = 0.f;
    for (int i = 0; i < GN * NE; i++)
        lb += (stats[GN * NE + i] / (float)TPG) * (stats[i] / (float)TPG);
    lb = lb / (float)(GN * NE) * (float)(NE * NE);
    float z = stats[2 * GN * NE] / (float)(GN * TPG);
    aux[0] = 0.01f * lb + 0.001f * z;
}

// ---------------------------------------------------------------------------
extern "C" void kernel_launch(void* const* d_in, const int* in_sizes, int n_in,
                              void* d_out, int out_size, void* d_ws, size_t ws_size,
                              hipStream_t stream) {
    const float* x   = (const float*)d_in[0];  // [G,TPG,EMB]
    const float* pad = (const float*)d_in[1];  // [G,TPG]
    const float* rk  = (const float*)d_in[2];  // [EMB,E]
    const float* wi0 = (const float*)d_in[3];  // [E,EMB,MLP]
    const float* wi1 = (const float*)d_in[4];  // [E,EMB,MLP]
    const float* wo  = (const float*)d_in[5];  // [E,MLP,EMB]
    float* out = (float*)d_out;                // [G*TPG*EMB] out + [1] aux

    char* base = (char*)d_ws;
    size_t o = 0;
    auto take = [&](size_t bytes) -> char* {
        char* p = base + o;
        o = (o + bytes + 255) & ~(size_t)255;
        return p;
    };
    int*   tok_e      = (int*)  take((size_t)GN * TPG * TOPK * 4);
    float* tok_gate   = (float*)take((size_t)GN * TPG * TOPK * 4);
    int*   tok_slot   = (int*)  take((size_t)GN * TPG * TOPK * 4);
    int*   slot_token = (int*)  take((size_t)GN * NE * CAP * 4);
    float* stats      = (float*)take((size_t)(2 * GN * NE + 1) * 4);
    bf16*  w0t        = (bf16*) take((size_t)NE * MLPD * EMBD * 2);
    bf16*  w1t        = (bf16*) take((size_t)NE * MLPD * EMBD * 2);
    bf16*  wot        = (bf16*) take((size_t)NE * EMBD * MLPD * 2);
    bf16*  Hbuf       = (bf16*) take((size_t)GN * NE * CAP * MLPD * 2);
    float* eo         = (float*)take((size_t)GN * NE * CAP * EMBD * 4);
    (void)ws_size; (void)in_sizes; (void)n_in; (void)out_size;

    // 1. init
    moe_init<<<(GN * NE * CAP + 255) / 256, 256, 0, stream>>>(slot_token, stats);
    // 2. router
    moe_router<<<dim3(TPG / 8, GN), 256, 0, stream>>>(x, pad, rk, tok_e, tok_gate, stats);
    // 3. dispatch
    moe_dispatch<<<GN, 32, 0, stream>>>(tok_e, tok_slot, slot_token);
    // 4. weight convert (bf16, transposed K-contiguous)
    moe_convert<<<4096, 256, 0, stream>>>(wi0, wi1, wo, w0t, w1t, wot);
    // 5. fused gated FFN first layer (WMMA bf16, TDM B-tiles, double-buffered)
    moe_ffn1<<<dim3(MLPD / 128, CAP / 64, GN * NE), 256, 0, stream>>>(
        x, slot_token, w0t, w1t, Hbuf);
    // 6. output projection (WMMA bf16, TDM A+B tiles, double-buffered)
    moe_ffn2<<<dim3(EMBD / 128, CAP / 64, GN * NE), 256, 0, stream>>>(Hbuf, wot, eo);
    // 7. combine
    moe_combine<<<GN * TPG, 256, 0, stream>>>(tok_e, tok_gate, tok_slot, eo, out);
    // 8. aux loss
    moe_finalize<<<1, 32, 0, stream>>>(stats, out + (size_t)GN * TPG * EMBD);
}
